// RPN_61211873902656
// MI455X (gfx1250) — compile-verified, matrix operand order checked by reading, not run on hardware
//
#include <hip/hip_runtime.h>
#include <hip/hip_bf16.h>

typedef __attribute__((ext_vector_type(16))) _Float16 v16h;
typedef __attribute__((ext_vector_type(8)))  _Float16 v8h;
typedef __attribute__((ext_vector_type(8)))  float    v8f;

#define BDIM 4
#define CIN 1280
#define HH 48
#define WW 48
#define HID 256
#define AA 9
#define MTOT (BDIM*HH*WW)      // 9216
#define KTOT (CIN*9)           // 11520
#define NPROP (AA*HH*WW)       // 20736
#define KKEEP 512

#define TILE_M 128
#define TILE_N 128
#define KC 32
#define LDA 40                 // padded row stride in halfs (80B, keeps 16B chunks aligned)
#define NSTEPS (KTOT / KC)     // 360

// ---------------- conversion / packing ----------------
// features NCHW f32 -> NHWC f16 : featT[((b*48+y)*48+x)*1280 + c]
__global__ void cvt_feat_kernel(const float* __restrict__ f, _Float16* __restrict__ o) {
    int t = blockIdx.x * blockDim.x + threadIdx.x;
    if (t >= BDIM * HH * WW * CIN) return;
    int c   = t % CIN;
    int pix = t / CIN;           // (b*48+y)*48+x
    int x   = pix % WW;
    int y   = (pix / WW) % HH;
    int b   = pix / (HH * WW);
    o[t] = (_Float16)f[(((size_t)b * CIN + c) * HH + y) * WW + x];
}

// W1 OIHW [256][1280][3][3] -> wpack2[n][K] with K = tap*1280 + ci, tap = ky*3+kx
__global__ void pack_w_kernel(const float* __restrict__ w1, _Float16* __restrict__ wp) {
    int t = blockIdx.x * blockDim.x + threadIdx.x;
    if (t >= HID * KTOT) return;
    int n   = t / KTOT;
    int r   = t - n * KTOT;
    int tap = r / CIN;
    int ci  = r - tap * CIN;
    wp[t] = (_Float16)w1[(size_t)n * KTOT + (size_t)ci * 9 + tap];
}

// ---------------- implicit-GEMM 3x3 conv via WMMA ----------------
__device__ __forceinline__ void stage_a(_Float16* __restrict__ As, int row, int kq,
                                        const _Float16* __restrict__ featT,
                                        int bb, int yy, int xx, int tap, int c0) {
    int ky = tap / 3;                 // uniform scalar
    int kx = tap - ky * 3;
    int sy = yy + ky - 1;
    int sx = xx + kx - 1;
    v8h lo = {}, hi = {};
    if ((unsigned)sy < (unsigned)HH && (unsigned)sx < (unsigned)WW) {
        const _Float16* p = featT + ((((size_t)bb * HH + sy) * WW + sx) * CIN + c0 + kq);
        lo = *(const v8h*)p;
        hi = *(const v8h*)(p + 8);
    }
    *(v8h*)(As + row * LDA + kq)     = lo;
    *(v8h*)(As + row * LDA + kq + 8) = hi;
}

__device__ __forceinline__ v16h load_frag_lds(const _Float16* p) {
    v8h lo = *(const v8h*)p;
    v8h hi = *(const v8h*)(p + 16);
    return __builtin_shufflevector(lo, hi, 0,1,2,3,4,5,6,7,8,9,10,11,12,13,14,15);
}

__global__ __launch_bounds__(256) void conv1_wmma_kernel(
    const _Float16* __restrict__ featT,    // [B][H][W][C] f16
    const _Float16* __restrict__ wpack2,   // [N=256][K=11520] f16
    const float*    __restrict__ b1,
    _Float16*       __restrict__ hid)      // [M][256] f16
{
    __shared__ _Float16 As[2][TILE_M * LDA];   // 2 x 10240 B

    const int tid  = threadIdx.x;
    const int lane = tid & 31;
    const int wave = tid >> 5;
    const int mtile = blockIdx.x;              // 72
    const int ntile = blockIdx.y;              // 2

    const int wm = (wave & 3) * 32;            // wave M offset (32 rows)
    const int wn = (wave >> 2) * 64;           // wave N offset (64 cols)

    // A-staging assignment: 2 threads per row, 16 halfs each
    const int m_st = tid >> 1;                 // 0..127
    const int kq   = (tid & 1) * 16;           // 0 or 16
    const int gm   = mtile * TILE_M + m_st;
    const int bb   = gm / (HH * WW);
    const int rem  = gm - bb * HH * WW;
    const int yy   = rem / WW;
    const int xx   = rem - yy * WW;

    // per-lane WMMA fragment coords (wave32, 16-bit 16x32):
    // lanes 0-15: row = lane, K = 0..7 & 16..23 ; lanes 16-31: row = lane-16, K = 8..15 & 24..31
    const int lr  = lane & 15;
    const int kb  = (lane >> 4) * 8;

    // B fragments direct from global: fixed column, contiguous k
    const _Float16* pB[4];
    #pragma unroll
    for (int n0 = 0; n0 < 4; ++n0) {
        int c = ntile * TILE_N + wn + n0 * 16 + lr;
        pB[n0] = wpack2 + (size_t)c * KTOT + kb;
    }

    v8f acc0[4] = {};
    v8f acc1[4] = {};

    // prologue: stage step 0
    stage_a(As[0], m_st, kq, featT, bb, yy, xx, /*tap=*/0, /*c0=*/0);

    for (int step = 0; step < NSTEPS; ++step) {
        const int tap = step / 40;             // uniform (SALU)
        const int c0  = (step - tap * 40) * KC;
        __syncthreads();                       // staged buffer (step&1) ready

        if (step + 1 < NSTEPS) {               // stage next chunk into other buffer
            int ns   = step + 1;
            int ntap = ns / 40;
            int nc0  = (ns - ntap * 40) * KC;
            stage_a(As[(step + 1) & 1], m_st, kq, featT, bb, yy, xx, ntap, nc0);
        }

        const _Float16* Ab = As[step & 1];
        const int k0 = tap * CIN + c0;

        v16h a0 = load_frag_lds(Ab + (wm + lr)      * LDA + kb);
        v16h a1 = load_frag_lds(Ab + (wm + 16 + lr) * LDA + kb);

        #pragma unroll
        for (int n0 = 0; n0 < 4; ++n0) {
            v8h blo = *(const v8h*)(pB[n0] + k0);
            v8h bhi = *(const v8h*)(pB[n0] + k0 + 16);
            v16h bf = __builtin_shufflevector(blo, bhi,
                        0,1,2,3,4,5,6,7,8,9,10,11,12,13,14,15);
            __builtin_prefetch((const void*)(pB[n0] + k0 + KC), 0, 1);
            acc0[n0] = __builtin_amdgcn_wmma_f32_16x16x32_f16(
                false, a0, false, bf, (short)0, acc0[n0], false, false);
            acc1[n0] = __builtin_amdgcn_wmma_f32_16x16x32_f16(
                false, a1, false, bf, (short)0, acc1[n0], false, false);
        }
    }

    // D layout: vgpr i -> M = i + 8*(lane>=16), N = lane&15 ; fuse bias + LeakyReLU
    #pragma unroll
    for (int n0 = 0; n0 < 4; ++n0) {
        const int n = ntile * TILE_N + wn + n0 * 16 + lr;
        const float bias = b1[n];
        #pragma unroll
        for (int i = 0; i < 8; ++i) {
            int mrow = mtile * TILE_M + wm + i + 8 * (lane >> 4);
            float v0 = acc0[n0][i] + bias;
            v0 = (v0 >= 0.0f) ? v0 : 0.01f * v0;
            hid[(size_t)mrow * HID + n] = (_Float16)v0;
            float v1 = acc1[n0][i] + bias;
            v1 = (v1 >= 0.0f) ? v1 : 0.01f * v1;
            hid[(size_t)(mrow + 16) * HID + n] = (_Float16)v1;
        }
    }
}

// ---------------- 1x1 head + box decode ----------------
__global__ void head_decode_kernel(
    const _Float16* __restrict__ hid,      // [M][256]
    const float* __restrict__ whead,       // [45][256]
    const float* __restrict__ bhead,       // [45]
    const float* __restrict__ anchor,      // [9][2]
    float* __restrict__ proposals,         // [B][NPROP][4]
    float* __restrict__ msk)               // [B][NPROP] masked scores
{
    int t = blockIdx.x * blockDim.x + threadIdx.x;
    if (t >= BDIM * AA * HH * WW) return;
    int x = t % WW;
    int y = (t / WW) % HH;
    int a = (t / (HH * WW)) % AA;
    int b = t / (AA * HH * WW);
    int m = (b * HH + y) * WW + x;
    const _Float16* hp = hid + (size_t)m * HID;

    float o[5];
    #pragma unroll
    for (int c5 = 0; c5 < 5; ++c5) {
        int oc = a * 5 + c5;
        const float* wp = whead + (size_t)oc * HID;
        float accv = bhead[oc];
        for (int k = 0; k < HID; k += 8) {
            v8h h = *(const v8h*)(hp + k);
            #pragma unroll
            for (int j = 0; j < 8; ++j) accv += (float)h[j] * wp[k + j];
        }
        o[c5] = accv;
    }
    float s  = 1.0f / (1.0f + expf(-o[0]));
    float wa = anchor[a * 2 + 0];
    float ha = anchor[a * 2 + 1];
    float pxc = ((float)x + 0.5f) + o[1] * wa;
    float pyc = ((float)y + 0.5f) + o[2] * ha;
    float pw  = wa * expf(o[3]);
    float ph  = ha * expf(o[4]);

    int n = (a * HH + y) * WW + x;
    size_t pb = ((size_t)b * NPROP + n) * 4;
    proposals[pb + 0] = pxc - 0.5f * pw;
    proposals[pb + 1] = pyc - 0.5f * ph;
    proposals[pb + 2] = pxc + 0.5f * pw;
    proposals[pb + 3] = pyc + 0.5f * ph;
    msk[(size_t)b * NPROP + n] = (s > 0.5f) ? s : -1.0f;
}

// ---------------- per-batch top-K + NMS ----------------
__global__ __launch_bounds__(512) void topk_nms_kernel(
    const float* __restrict__ proposals,
    float* __restrict__ msk,               // destroyed (regenerated each launch)
    float* __restrict__ out)               // boxes | scores | keep
{
    const int b = blockIdx.x;
    const int tid = threadIdx.x;
    __shared__ float rv[512];
    __shared__ int   ri[512];
    __shared__ float sbx[512 * 4];
    __shared__ float sts[512];
    __shared__ int   ssup[512];
    __shared__ int   sidx[512];

    float* ms = msk + (size_t)b * NPROP;

    // iterative argmax: larger score wins, tie -> smaller index (matches lax.top_k)
    for (int it = 0; it < KKEEP; ++it) {
        float best = -3.0e38f; int bi = 0x7fffffff;
        for (int j = tid; j < NPROP; j += 512) {
            float v = ms[j];
            if (v > best) { best = v; bi = j; }
        }
        rv[tid] = best; ri[tid] = bi;
        __syncthreads();
        for (int off = 256; off > 0; off >>= 1) {
            if (tid < off) {
                float v2 = rv[tid + off]; int i2 = ri[tid + off];
                if (v2 > rv[tid] || (v2 == rv[tid] && i2 < ri[tid])) { rv[tid] = v2; ri[tid] = i2; }
            }
            __syncthreads();
        }
        if (tid == 0) { sts[it] = rv[0]; sidx[it] = ri[0]; ms[ri[0]] = -3.4e38f; }
        __syncthreads();
    }

    // gather boxes
    int idx = sidx[tid];
    const float* pp = proposals + ((size_t)b * NPROP + idx) * 4;
    float x1 = pp[0], y1 = pp[1], x2 = pp[2], y2 = pp[3];
    sbx[tid * 4 + 0] = x1; sbx[tid * 4 + 1] = y1;
    sbx[tid * 4 + 2] = x2; sbx[tid * 4 + 3] = y2;
    ssup[tid] = 0;
    float myts = sts[tid];
    __syncthreads();

    float area_j = (x2 - x1) * (y2 - y1);
    int mykeep = 0;
    for (int i = 0; i < KKEEP; ++i) {
        int   sup_i = ssup[i];
        float ts_i  = sts[i];
        float bx1 = sbx[i * 4 + 0], by1 = sbx[i * 4 + 1];
        float bx2 = sbx[i * 4 + 2], by2 = sbx[i * 4 + 3];
        __syncthreads();
        int ki = (ts_i > 0.5f) && !sup_i;
        if (tid == i) mykeep = ki;
        if (ki) {
            float ix1 = fmaxf(bx1, x1), iy1 = fmaxf(by1, y1);
            float ix2 = fminf(bx2, x2), iy2 = fminf(by2, y2);
            float inter = fmaxf(ix2 - ix1, 0.0f) * fmaxf(iy2 - iy1, 0.0f);
            float area_i = (bx2 - bx1) * (by2 - by1);
            float uni = fmaxf(area_i + area_j - inter, 1e-8f);
            if (inter / uni > 0.7f) ssup[tid] = 1;
        }
        __syncthreads();
    }

    float kf = mykeep ? 1.0f : 0.0f;
    float* boxes_out  = out;
    float* scores_out = out + (size_t)BDIM * KKEEP * 4;
    float* keep_out   = scores_out + (size_t)BDIM * KKEEP;
    size_t ob = (size_t)b * KKEEP + tid;
    boxes_out[ob * 4 + 0] = x1 * kf;
    boxes_out[ob * 4 + 1] = y1 * kf;
    boxes_out[ob * 4 + 2] = x2 * kf;
    boxes_out[ob * 4 + 3] = y2 * kf;
    scores_out[ob] = myts * kf;
    keep_out[ob]   = kf;
}

// ---------------- host launcher ----------------
extern "C" void kernel_launch(void* const* d_in, const int* in_sizes, int n_in,
                              void* d_out, int out_size, void* d_ws, size_t ws_size,
                              hipStream_t stream) {
    const float* features = (const float*)d_in[0];
    const float* W1       = (const float*)d_in[1];
    const float* b1       = (const float*)d_in[2];
    const float* W_head   = (const float*)d_in[3];
    const float* b_head   = (const float*)d_in[4];
    const float* anchor   = (const float*)d_in[5];

    char* ws = (char*)d_ws;
    size_t off = 0;
    _Float16* featT  = (_Float16*)(ws + off); off += (size_t)BDIM * HH * WW * CIN * 2;   // 23.6 MB
    _Float16* wpack2 = (_Float16*)(ws + off); off += (size_t)HID * KTOT * 2;             //  5.9 MB
    _Float16* hid    = (_Float16*)(ws + off); off += (size_t)MTOT * HID * 2;             //  4.7 MB
    float* proposals = (float*)(ws + off);    off += (size_t)BDIM * NPROP * 4 * 4;       //  1.3 MB
    float* msk       = (float*)(ws + off);    off += (size_t)BDIM * NPROP * 4;           //  0.3 MB

    int nfeat = BDIM * HH * WW * CIN;
    cvt_feat_kernel<<<(nfeat + 255) / 256, 256, 0, stream>>>(features, featT);

    int nw = HID * KTOT;
    pack_w_kernel<<<(nw + 255) / 256, 256, 0, stream>>>(W1, wpack2);

    dim3 g1(MTOT / TILE_M, HID / TILE_N);   // (72, 2)
    conv1_wmma_kernel<<<g1, 256, 0, stream>>>(featT, wpack2, b1, hid);

    int nhd = BDIM * AA * HH * WW;
    head_decode_kernel<<<(nhd + 255) / 256, 256, 0, stream>>>(hid, W_head, b_head, anchor,
                                                              proposals, msk);

    topk_nms_kernel<<<BDIM, 512, 0, stream>>>(proposals, msk, (float*)d_out);
}
